// SpikingConv2DLayer_11381663334463
// MI455X (gfx1250) — compile-verified, matrix operand order checked by reading, not run on hardware
//
#include <hip/hip_runtime.h>

// ---------------------------------------------------------------------------
// CDNA5 (gfx1250) fused spiking-conv forward.
//   kernel 1: prep      - bf16 hi/lo split of weights + Gram matrix, inv_norm
//   kernel 2: scan      - fused conv + recurrent scan, WMMA bf16, TDM prefetch
//   kernel 3: finalize  - loss / spk_spread scalars
// ---------------------------------------------------------------------------

typedef __bf16 bf16_t;
typedef __attribute__((ext_vector_type(16))) __bf16       v16bf;
typedef __attribute__((ext_vector_type(8)))  float        v8f;
typedef __attribute__((ext_vector_type(4)))  unsigned int v4u;
typedef __attribute__((ext_vector_type(8)))  int          v8i;
typedef __attribute__((ext_vector_type(4)))  int          v4i;

#define BB     32
#define CIN_   8
#define TT     256
#define WW     128
#define COUT_  64
#define OUTW   128
#define KPAD   256        // 40 (cin,kt) groups * 6 slots, padded to 8 K-chunks of 32
#define NGROUP 40
#define GSLOT  6          // 5 taps + 1 zero-weight slot so K-pairs never straddle taps
#define WSLAB  16         // w columns per workgroup
#define ROWW   24         // bf16 row width (16 + 4 halo + pad), even
#define ROWDW  12         // dwords per row
#define OUT_ELEMS 67108864ull   // 32*64*256*128

__device__ __forceinline__ v8f wmma_bf16(v16bf a, v16bf b, v8f c) {
  // D = A(16x32 bf16) * B(32x16 bf16) + C(16x16 f32)
  return __builtin_amdgcn_wmma_f32_16x16x32_bf16(false, a, false, b, (short)0, c,
                                                 false, false);
}

// ---- Tensor Data Mover: 2D tile (dim0 x dim1 f32, row stride stride0) -> LDS.
// D# built per CDNA5 ISA ch.8 (group0 128b, group1 256b); groups 2/3 zero (<=2D).
// This toolchain exposes the 6-arg builtin: (v4u, v8i, v4i, v4i, v8i, i32 cpol).
__device__ __forceinline__ void tdm_load_2d(unsigned lds_off,
                                            unsigned long long gaddr,
                                            int dim0, int dim1,
                                            long long stride0) {
  v4u g0; v8i g1; v4i z4; v8i z8;
  z4[0] = 0; z4[1] = 0; z4[2] = 0; z4[3] = 0;
  z8[0] = 0; z8[1] = 0; z8[2] = 0; z8[3] = 0;
  z8[4] = 0; z8[5] = 0; z8[6] = 0; z8[7] = 0;
  g0[0] = 1u;                                              // count=1 valid D#
  g0[1] = lds_off;                                         // lds_addr (bytes)
  g0[2] = (unsigned)(gaddr & 0xffffffffull);               // global_addr[31:0]
  g0[3] = (unsigned)((gaddr >> 32) & 0x1ffffffull)         // global_addr[56:32]
        | (2u << 30);                                      // type = 2 (image)
  g1[0] = (int)(2u << 16);                                 // data_size=2 (4B)
  g1[1] = (int)(((unsigned)dim0 & 0xffffu) << 16);         // tensor_dim0 lo16
  g1[2] = (int)((((unsigned)dim0 >> 16) & 0xffffu)         // tensor_dim0 hi16
        | (((unsigned)dim1 & 0xffffu) << 16));             // tensor_dim1 lo16
  g1[3] = (int)(((unsigned)dim0 & 0xffffu) << 16);         // tile_dim0
  g1[4] = (int)((unsigned)dim1 & 0xffffu);                 // tile_dim1
  g1[5] = (int)(stride0 & 0xffffffffll);                   // dim0_stride lo32
  g1[6] = (int)((stride0 >> 32) & 0xffffll);               // dim0_stride hi16
  g1[7] = 0;
  __builtin_amdgcn_tensor_load_to_lds(g0, g1, z4, z4, z8, 0);
}

// ---------------------------------------------------------------------------
// Kernel 1: prep
// ---------------------------------------------------------------------------
__global__ __launch_bounds__(256) void prep_kernel(
    const float* __restrict__ w,
    bf16_t* __restrict__ wt_hi, bf16_t* __restrict__ wt_lo,
    bf16_t* __restrict__ dt_hi, bf16_t* __restrict__ dt_lo,
    float* __restrict__ inv_norm, float* __restrict__ g_tsum,
    float* __restrict__ g_loss) {
  const int tid = threadIdx.x;
  // conv weights -> N-major B layout with padded K (zero slots)
  for (int idx = tid; idx < COUT_ * KPAD; idx += 256) {
    const int o = idx >> 8, kp = idx & 255;
    const int g = kp / GSLOT, slot = kp - g * GSLOT;
    float v = 0.f;
    if (kp < NGROUP * GSLOT && slot < 5) v = w[o * 200 + g * 5 + slot];
    const bf16_t h = (bf16_t)v;
    wt_hi[idx] = h;
    wt_lo[idx] = (bf16_t)(v - (float)h);
  }
  // Gram matrix d[k][n] (symmetric), N-major, hi/lo split
  for (int idx = tid; idx < COUT_ * COUT_; idx += 256) {
    const int n = idx >> 6, k = idx & 63;
    float acc = 0.f;
    for (int q = 0; q < 200; ++q) acc += w[k * 200 + q] * w[n * 200 + q];
    const bf16_t h = (bf16_t)acc;
    dt_hi[idx] = h;
    dt_lo[idx] = (bf16_t)(acc - (float)h);
  }
  if (tid < COUT_) {
    float nrm = 0.f;
    for (int q = 0; q < 200; ++q) { const float t = w[tid * 200 + q]; nrm += t * t; }
    inv_norm[tid] = 1.f / (nrm + 1e-8f);
  }
  g_tsum[tid] = 0.f;
  if (tid == 0) *g_loss = 0.f;
}

// ---------------------------------------------------------------------------
// Kernel 2: fused conv + scan.  grid (8 wslabs, 32 batches), 128 threads.
// Wave n owns output-channel tile n (16 couts) for the 16-wide w slab.
// ---------------------------------------------------------------------------
__global__ __launch_bounds__(128) void spiking_scan_kernel(
    const float* __restrict__ x,
    const bf16_t* __restrict__ wt_hi, const bf16_t* __restrict__ wt_lo,
    const bf16_t* __restrict__ dt_hi, const bf16_t* __restrict__ dt_lo,
    const float* __restrict__ inv_norm_g, const float* __restrict__ bias_g,
    const float* __restrict__ beta_g,
    float* __restrict__ out, float* __restrict__ g_tsum,
    float* __restrict__ g_loss) {
  // rolling bf16 x-window: hi/lo + one-element-shifted copies (aligned K pairs)
  __shared__ __attribute__((aligned(16))) bf16_t rh [NGROUP][ROWW];
  __shared__ __attribute__((aligned(16))) bf16_t rhs[NGROUP][ROWW];
  __shared__ __attribute__((aligned(16))) bf16_t rl [NGROUP][ROWW];
  __shared__ __attribute__((aligned(16))) bf16_t rls[NGROUP][ROWW];
  __shared__ __attribute__((aligned(16))) float  stage[2][CIN_ * ROWW]; // TDM dst
  __shared__ __attribute__((aligned(16))) bf16_t spk[WSLAB][COUT_];
  __shared__ __attribute__((aligned(16))) float  outst[COUT_][WSLAB];
  __shared__ float tsum_l[TT];
  __shared__ float loss_l;

  const int tid  = threadIdx.x;
  const int lane = tid & 31;
  const int wave = tid >> 5;          // = cout tile
  const int half = lane >> 4;
  const int m16  = lane & 15;         // A-row M = w within slab / B-row N
  const int b    = blockIdx.y;
  const int w0   = blockIdx.x * WSLAB;

  const int lo  = (w0 - 2 > 0) ? (w0 - 2) : 0;
  const int hi  = (w0 + ROWW - 2 < WW) ? (w0 + ROWW - 2) : WW;
  const int cnt = hi - lo;            // valid strip width (<= 24)

  const float beta   = beta_g[0];
  const float ombeta = 1.f - beta;
  const int   c_lane = wave * 16 + m16;     // this lane's output channel
  const float invn   = inv_norm_g[c_lane];
  const float bias   = bias_g[c_lane];

  // ---- init LDS
  for (int i = tid; i < NGROUP * ROWW; i += 128) {
    const int g = i / ROWW, j = i - g * ROWW;
    rh[g][j] = (bf16_t)0.f; rhs[g][j] = (bf16_t)0.f;
    rl[g][j] = (bf16_t)0.f; rls[g][j] = (bf16_t)0.f;
  }
  for (int i = tid; i < WSLAB * COUT_; i += 128) (&spk[0][0])[i] = (bf16_t)0.f;
  for (int i = tid; i < TT; i += 128) tsum_l[i] = 0.f;
  if (tid == 0) loss_l = 0.f;
  __syncthreads();

  // ---- prologue: trow 0,1 direct from global into ring slots 0,1
  for (int r = 0; r < 2; ++r) {
    for (int i = tid; i < CIN_ * ROWW; i += 128) {
      const int cin = i / ROWW, j = i - cin * ROWW;
      const int wg_ = w0 + j - 2;
      float v = (wg_ >= 0 && wg_ < WW)
                    ? x[((size_t)(b * CIN_ + cin) * TT + r) * WW + wg_] : 0.f;
      const bf16_t h = (bf16_t)v;
      const bf16_t l2 = (bf16_t)(v - (float)h);
      const int row = cin * 5 + r;
      rh[row][j] = h; rl[row][j] = l2;
      if (j > 0) { rhs[row][j - 1] = h; rls[row][j - 1] = l2; }
    }
  }
  // prefetch trow=2 asynchronously (TDM, EXEC-independent; wave0 only)
  if (wave == 0) {
    unsigned long long ga = (unsigned long long)(size_t)(
        x + ((size_t)(b * CIN_) * TT + 2) * WW + lo);
    tdm_load_2d((unsigned)(size_t)&stage[0][0], ga, cnt, CIN_,
                (long long)(TT * WW));
  }

  v8f mem, prev;
#pragma unroll
  for (int r = 0; r < 8; ++r) { mem[r] = 0.f; prev[r] = 0.f; }
  float loss_acc = 0.f;

  const int baseA = m16 >> 1;
  const unsigned int* ph = (m16 & 1) ? (const unsigned int*)&rhs[0][0]
                                     : (const unsigned int*)&rh[0][0];
  const unsigned int* pl = (m16 & 1) ? (const unsigned int*)&rls[0][0]
                                     : (const unsigned int*)&rl[0][0];

  for (int t = 0; t < TT; ++t) {
    // -- issue TDM for trow t+3 (double buffered), guarantee trow t+2 landed
    if (wave == 0) {
      if (t + 3 < TT) {
        unsigned long long ga = (unsigned long long)(size_t)(
            x + ((size_t)(b * CIN_) * TT + (t + 3)) * WW + lo);
        tdm_load_2d((unsigned)(size_t)&stage[(t + 3) & 1][0], ga, cnt, CIN_,
                    (long long)(TT * WW));
        __builtin_amdgcn_s_wait_tensorcnt(1);
      } else {
        __builtin_amdgcn_s_wait_tensorcnt(0);
      }
    }
    __syncthreads();  // (a) stage[(t+2)&1] visible; prior ring-slot reads done

    // -- convert trow t+2 into ring slot (t+2)%5 (hi/lo + shifted copies)
    {
      const int trow = t + 2;
      const int slot = trow % 5;
      const float* sg = &stage[trow & 1][0];
      for (int i = tid; i < CIN_ * ROWW; i += 128) {
        const int cin = i / ROWW, j = i - cin * ROWW;
        const int wg_ = w0 + j - 2;
        float v = 0.f;
        if (trow < TT && wg_ >= lo && wg_ < hi) v = sg[cin * cnt + (wg_ - lo)];
        const bf16_t h = (bf16_t)v;
        const bf16_t l2 = (bf16_t)(v - (float)h);
        const int row = cin * 5 + slot;
        rh[row][j] = h; rl[row][j] = l2;
        if (j > 0) { rhs[row][j - 1] = h; rls[row][j - 1] = l2; }
      }
    }
    __syncthreads();  // (b) ring + spk(t-1) ready

    // -- conv: C(16w x 16cout) += im2col(x) * W    (hi/lo split, 3 products)
    v8f cacc, racc;
#pragma unroll
    for (int r = 0; r < 8; ++r) { cacc[r] = 0.f; racc[r] = 0.f; }
    const int s0 = (t + 3) % 5;  // ring slot of kt=0 (uniform)
#pragma unroll
    for (int chunk = 0; chunk < 8; ++chunk) {
      union { v16bf v; unsigned int u[8]; } Ah, Al;
#pragma unroll
      for (int i = 0; i < 8; ++i) {
        // K-pair index m = chunk*16 + half*4 + d: only two variants per (chunk,i);
        // everything except the uniform ring rotation s0 folds to constants, so
        // both offsets are scalar (SALU) and combine with one cndmask + add.
        const int d  = (i < 4 ? i : i + 4);
        const int mA = chunk * 16 + d;       // half = 0
        const int mB = mA + 4;               // half = 1
        int gA = mA / 3, mmA = mA - gA * 3;
        if (gA >= NGROUP) { gA = NGROUP - 1; mmA = 0; }   // padded K: B is zero
        int gB = mB / 3, mmB = mB - gB * 3;
        if (gB >= NGROUP) { gB = NGROUP - 1; mmB = 0; }
        const int cinA = gA / 5, ktA = gA - (gA / 5) * 5;
        const int cinB = gB / 5, ktB = gB - (gB / 5) * 5;
        int slotA = s0 + ktA; if (slotA >= 5) slotA -= 5;  // uniform
        int slotB = s0 + ktB; if (slotB >= 5) slotB -= 5;  // uniform
        const int offA = (cinA * 5 + slotA) * ROWDW + mmA; // uniform (SGPR)
        const int offB = (cinB * 5 + slotB) * ROWDW + mmB; // uniform (SGPR)
        const int off  = baseA + (half ? offB : offA);
        Ah.u[i] = ph[off];
        Al.u[i] = pl[off];
      }
      const v4u* bh = (const v4u*)(wt_hi + (size_t)c_lane * KPAD + chunk * 32 + half * 8);
      const v4u* bl = (const v4u*)(wt_lo + (size_t)c_lane * KPAD + chunk * 32 + half * 8);
      union { v16bf v; v4u q[2]; } Bh, Bl;
      Bh.q[0] = bh[0]; Bh.q[1] = bh[2];
      Bl.q[0] = bl[0]; Bl.q[1] = bl[2];
      cacc = wmma_bf16(Ah.v, Bh.v, cacc);
      cacc = wmma_bf16(Al.v, Bh.v, cacc);
      cacc = wmma_bf16(Ah.v, Bl.v, cacc);
    }

    // -- rst = spk(t-1) @ d   (spk exact in bf16; d hi/lo split)
    if (t > 0) {
#pragma unroll
      for (int chunk = 0; chunk < 2; ++chunk) {
        const int kb = chunk * 32 + half * 8;
        const v4u* sp = (const v4u*)&spk[m16][0];
        union { v16bf v; v4u q[2]; } Aa, Bh, Bl;
        Aa.q[0] = sp[kb >> 3];
        Aa.q[1] = sp[(kb + 16) >> 3];
        const v4u* dh = (const v4u*)(dt_hi + (size_t)c_lane * COUT_ + kb);
        const v4u* dl = (const v4u*)(dt_lo + (size_t)c_lane * COUT_ + kb);
        Bh.q[0] = dh[0]; Bh.q[1] = dh[2];
        Bl.q[0] = dl[0]; Bl.q[1] = dl[2];
        racc = wmma_bf16(Aa.v, Bh.v, racc);
        racc = wmma_bf16(Aa.v, Bl.v, racc);
      }
    }

    // -- membrane update + spike generation (C-fragment: M=r+8*half, N=m16)
    v8f sfrag;
    float s8 = 0.f;
#pragma unroll
    for (int r = 0; r < 8; ++r) {
      const float m_ = (mem[r] - racc[r]) * beta + cacc[r] * ombeta;
      mem[r] = m_;
      const float mthr = m_ * invn - bias;
      const float s = (mthr > 0.f) ? (3.0f - prev[r]) : 0.f;
      prev[r] += s;
      sfrag[r] = s;
      loss_acc += s * s;
      s8 += s;
    }
    atomicAdd(&tsum_l[t], s8);

    __syncthreads();  // (c) all reads of spk(t-1)/outst done
#pragma unroll
    for (int r = 0; r < 8; ++r) {
      const int wl = r + 8 * half;
      spk[wl][c_lane] = (bf16_t)sfrag[r];
      outst[c_lane][wl] = sfrag[r];
    }
    __syncthreads();  // (d) spk(t)/outst visible

    // -- coalesced output: out[b][c][t][w0..w0+15]
    {
      const int c  = tid >> 1;
      const int h8 = (tid & 1) * 8;
      const float4 v0 = *(const float4*)&outst[c][h8];
      const float4 v1 = *(const float4*)&outst[c][h8 + 4];
      float* dst = out + (((size_t)(b * COUT_ + c) * TT + t) * OUTW + w0 + h8);
      *(float4*)dst = v0;
      *(float4*)(dst + 4) = v1;
    }
  }

  // ---- reductions
  atomicAdd(&loss_l, loss_acc);
  __syncthreads();
  if (tid == 0) atomicAdd(g_loss, loss_l);
  for (int i = tid; i < TT; i += 128) atomicAdd(&g_tsum[i], tsum_l[i]);
}

// ---------------------------------------------------------------------------
// Kernel 3: finalize scalars
// ---------------------------------------------------------------------------
__global__ __launch_bounds__(256) void finalize_kernel(
    const float* __restrict__ g_tsum, const float* __restrict__ g_loss,
    float* __restrict__ out_tail) {
  __shared__ float red[256];
  const int tid = threadIdx.x;
  red[tid] = g_tsum[tid];
  __syncthreads();
  for (int s2 = 128; s2 > 0; s2 >>= 1) {
    if (tid < s2) red[tid] = fmaxf(red[tid], red[tid + s2]);
    __syncthreads();
  }
  if (tid == 0) {
    out_tail[0] = 0.5f * (g_loss[0] / (float)OUT_ELEMS);  // loss
    out_tail[1] = red[0] / 262144.f;                      // spk_spread
  }
}

// ---------------------------------------------------------------------------
extern "C" void kernel_launch(void* const* d_in, const int* in_sizes, int n_in,
                              void* d_out, int out_size, void* d_ws,
                              size_t ws_size, hipStream_t stream) {
  const float* x    = (const float*)d_in[0];  // (32,8,256,128)
  const float* w    = (const float*)d_in[1];  // (64,8,5,5)
  const float* bvec = (const float*)d_in[2];  // (64,)
  const float* beta = (const float*)d_in[3];  // (1,)
  // d_in[4] = sigma: unused in forward (surrogate affects only backward)
  float* out = (float*)d_out;

  char* ws = (char*)d_ws;  // ~84 KB total
  bf16_t* wt_hi = (bf16_t*)ws;  ws += COUT_ * KPAD * sizeof(bf16_t);
  bf16_t* wt_lo = (bf16_t*)ws;  ws += COUT_ * KPAD * sizeof(bf16_t);
  bf16_t* dt_hi = (bf16_t*)ws;  ws += COUT_ * COUT_ * sizeof(bf16_t);
  bf16_t* dt_lo = (bf16_t*)ws;  ws += COUT_ * COUT_ * sizeof(bf16_t);
  float* inv_norm = (float*)ws; ws += COUT_ * sizeof(float);
  float* g_tsum   = (float*)ws; ws += TT * sizeof(float);
  float* g_loss   = (float*)ws; ws += 256;

  prep_kernel<<<1, 256, 0, stream>>>(w, wt_hi, wt_lo, dt_hi, dt_lo, inv_norm,
                                     g_tsum, g_loss);
  dim3 grid(OUTW / WSLAB, BB);  // 8 w-slabs x 32 batches = 256 workgroups
  spiking_scan_kernel<<<grid, 128, 0, stream>>>(x, wt_hi, wt_lo, dt_hi, dt_lo,
                                                inv_norm, bvec, beta, out,
                                                g_tsum, g_loss);
  finalize_kernel<<<1, 256, 0, stream>>>(g_tsum, g_loss, out + OUT_ELEMS);
}
